// BinLinear_48309792145693
// MI455X (gfx1250) — matrix operand order repaired and checked by reading, not compiled, not run on hardware
//
#include <hip/hip_runtime.h>

typedef _Float16 h16;
typedef __attribute__((ext_vector_type(16))) _Float16 v16h;
typedef __attribute__((ext_vector_type(8)))  _Float16 v8h;
typedef __attribute__((ext_vector_type(8)))  float    v8f;

#define NROWS 8192
#define DIN   4096
#define DOUT  4096
#define BN_EPS 1e-5f

// ---------------- Pass 1a: per-column partial sums over 256-row chunks ----------------
__global__ __launch_bounds__(256) void k_colstat_partial(const float* __restrict__ x,
                                                         float* __restrict__ psum,
                                                         float* __restrict__ psq) {
  const int j  = blockIdx.x * 256 + threadIdx.x;   // column
  const int r0 = blockIdx.y * 256;                 // row chunk
  const float* p = x + (size_t)r0 * DIN + j;
  float s = 0.f, q = 0.f;
#pragma unroll 8
  for (int r = 0; r < 256; ++r) {
    float v = p[(size_t)r * DIN];
    s += v;
    q += v * v;
  }
  psum[blockIdx.y * DIN + j] = s;
  psq [blockIdx.y * DIN + j] = q;
}

// ---------------- Pass 1b: finalize BN scale/bias per column ----------------
__global__ __launch_bounds__(256) void k_colstat_final(const float* __restrict__ psum,
                                                       const float* __restrict__ psq,
                                                       const float* __restrict__ gamma,
                                                       const float* __restrict__ bbeta,
                                                       float* __restrict__ colScale,
                                                       float* __restrict__ colBias) {
  const int j = blockIdx.x * 256 + threadIdx.x;
  float s = 0.f, q = 0.f;
#pragma unroll
  for (int c = 0; c < 32; ++c) {
    s += psum[c * DIN + j];
    q += psq [c * DIN + j];
  }
  const float mean = s * (1.f / NROWS);
  const float var  = q * (1.f / NROWS) - mean * mean;
  const float rstd = rsqrtf(var + BN_EPS);
  const float sc   = gamma[j] * rstd;
  colScale[j] = sc;
  colBias[j]  = bbeta[j] - mean * sc;
}

// ---------------- Pass 2: normalize, binarize to f16 {-1,0,+1}, row beta ----------------
__global__ __launch_bounds__(256) void k_binarize(const float* __restrict__ x,
                                                  const float* __restrict__ colScale,
                                                  const float* __restrict__ colBias,
                                                  h16* __restrict__ hb,
                                                  float* __restrict__ rowBeta) {
  __shared__ float red[256];
  const int r = blockIdx.x;
  const float4* xr  = (const float4*)(x + (size_t)r * DIN);
  const float4* scv = (const float4*)colScale;
  const float4* biv = (const float4*)colBias;
  unsigned long long* ob = (unsigned long long*)(hb + (size_t)r * DIN);
  float asum = 0.f;
#pragma unroll
  for (int it = 0; it < 4; ++it) {
    const int j4 = it * 256 + threadIdx.x;
    const float4 xv = xr[j4];
    const float4 sc = scv[j4];
    const float4 bi = biv[j4];
    const float h0 = fmaf(xv.x, sc.x, bi.x);
    const float h1 = fmaf(xv.y, sc.y, bi.y);
    const float h2 = fmaf(xv.z, sc.z, bi.z);
    const float h3 = fmaf(xv.w, sc.w, bi.w);
    asum += fabsf(h0) + fabsf(h1) + fabsf(h2) + fabsf(h3);
    union { h16 h[4]; unsigned long long u; } o;
    o.h[0] = (h16)((h0 > 0.f) ? 1.f : ((h0 < 0.f) ? -1.f : 0.f));
    o.h[1] = (h16)((h1 > 0.f) ? 1.f : ((h1 < 0.f) ? -1.f : 0.f));
    o.h[2] = (h16)((h2 > 0.f) ? 1.f : ((h2 < 0.f) ? -1.f : 0.f));
    o.h[3] = (h16)((h3 > 0.f) ? 1.f : ((h3 < 0.f) ? -1.f : 0.f));
    ob[j4] = o.u;
  }
  red[threadIdx.x] = asum;
  __syncthreads();
  for (int s = 128; s > 0; s >>= 1) {
    if (threadIdx.x < s) red[threadIdx.x] += red[threadIdx.x + s];
    __syncthreads();
  }
  if (threadIdx.x == 0) rowBeta[r] = red[0] * (1.f / DIN);
}

// ---------------- Pass 3: W fp32 -> f16 ----------------
__global__ __launch_bounds__(256) void k_w2h(const float* __restrict__ W,
                                             h16* __restrict__ Wh) {
  const size_t i = ((size_t)blockIdx.x * 256 + threadIdx.x) * 4;
  const float4 w = *(const float4*)(W + i);
  union { h16 h[4]; unsigned long long u; } o;
  o.h[0] = (h16)w.x; o.h[1] = (h16)w.y; o.h[2] = (h16)w.z; o.h[3] = (h16)w.w;
  *(unsigned long long*)(Wh + i) = o.u;
}

// ---------------- CDNA5 async LDS fill helpers ----------------
__device__ __forceinline__ void async_fill_b128(unsigned lds_byte_off, const void* gaddr) {
  // VDST = LDS byte address, VADDR = 64-bit global address (GV mode)
  asm volatile("global_load_async_to_lds_b128 %0, %1, off"
               :: "v"(lds_byte_off), "v"(gaddr) : "memory");
}
__device__ __forceinline__ void wait_async0() {
  asm volatile("s_wait_asynccnt 0" ::: "memory");
}

// ---------------- Pass 4: WMMA GEMM  out[b,o] = (hb . Wh^T) * alpha[o] * beta[b] ----------------
#define TM 128
#define TN 256
#define KT 32
#define LDT 40                    // f16 stride of an LDS tile row (32 + 8 pad, 16B aligned,
                                  // 80B row stride -> 20-bank step, conflict-free ds_load_b128)
#define KSTEPS (DIN / KT)         // 128

__global__ __launch_bounds__(256) void k_bin_gemm(const h16* __restrict__ A,    // [NROWS, DIN]  ±1 f16
                                                  const h16* __restrict__ Bw,   // [DOUT,  DIN]  W in f16
                                                  const float* __restrict__ alpha,
                                                  const float* __restrict__ rowBeta,
                                                  float* __restrict__ out) {
  __shared__ __align__(16) h16 smA[2][TM * LDT];   // 2 x 10 KB
  __shared__ __align__(16) h16 smB[2][TN * LDT];   // 2 x 20 KB   (60 KB total)

  const int tid  = threadIdx.x;
  const int lane = tid & 31;
  const int wave = tid >> 5;            // 0..7
  const int wm = (wave & 1) * 64;       // wave M offset (2 waves along M)
  const int wn = (wave >> 1) * 64;      // wave N offset (4 waves along N)
  const int mb0 = blockIdx.y * TM;
  const int nb0 = blockIdx.x * TN;

  // Tile fill: 16B chunks, 4 chunks per row of 32 f16.
  // A: 512 chunks (2/thread), B: 1024 chunks (4/thread).
  const int crow = tid >> 2;            // 0..63
  const int ckof = (tid & 3) * 8;       // f16 offset: 0,8,16,24

  const h16* gA[2];
  const h16* gB[4];
#pragma unroll
  for (int s = 0; s < 2; ++s) gA[s] = A  + (size_t)(mb0 + crow + 64 * s) * DIN + ckof;
#pragma unroll
  for (int s = 0; s < 4; ++s) gB[s] = Bw + (size_t)(nb0 + crow + 64 * s) * DIN + ckof;

  // Fragment addressing (wave32 16-bit WMMA layouts):
  // A 16x32: lane L holds row M=L%16, K in {kh..kh+7} U {kh+16..kh+23}, kh = (L/16)*8
  // B 32x16: lane L holds col N=L%16, K in {kb..kb+7} U {kb+16..kb+23}, kb = (L/16)*8
  const int lh     = lane & 15;
  const int khalfA = (lane >> 4) * 8;
  const int khalfB = (lane >> 4) * 8;

  v8f acc[4][4] = {};

  // Issue async fills of buffer `buf` with K-slice `ks`.
  auto fill = [&](int buf, int ks) {
    const size_t kof = (size_t)ks * KT;
#pragma unroll
    for (int s = 0; s < 2; ++s) {
      unsigned loff = (unsigned)(size_t)&smA[buf][(crow + 64 * s) * LDT + ckof];
      async_fill_b128(loff, (const void*)(gA[s] + kof));
    }
#pragma unroll
    for (int s = 0; s < 4; ++s) {
      unsigned loff = (unsigned)(size_t)&smB[buf][(crow + 64 * s) * LDT + ckof];
      async_fill_b128(loff, (const void*)(gB[s] + kof));
    }
  };

  fill(0, 0);
  wait_async0();
  __syncthreads();

  for (int ks = 0; ks < KSTEPS; ++ks) {
    const int cur = ks & 1;
    const int nxt = cur ^ 1;

    if (ks + 1 < KSTEPS) fill(nxt, ks + 1);     // async DMA into the other buffer
    if (ks + 2 < KSTEPS) {                      // global_prefetch_b8 two slices ahead
      const size_t kof2 = (size_t)(ks + 2) * KT;
      __builtin_prefetch((const void*)(gA[0] + kof2), 0, 2);
      __builtin_prefetch((const void*)(gB[0] + kof2), 0, 2);
    }

    // Load fragments from LDS in WMMA register layout.
    v16h afr[4], bfr[4];
#pragma unroll
    for (int i = 0; i < 4; ++i) {
      const h16* p = &smA[cur][(wm + i * 16 + lh) * LDT + khalfA];
      union { v16h v; v8h h[2]; } u;
      u.h[0] = *(const v8h*)(p);          // K = kh .. kh+7
      u.h[1] = *(const v8h*)(p + 16);     // K = kh+16 .. kh+23
      afr[i] = u.v;
    }
#pragma unroll
    for (int j = 0; j < 4; ++j) {
      const h16* p = &smB[cur][(wn + j * 16 + lh) * LDT + khalfB];
      union { v16h v; v8h h[2]; } u;
      u.h[0] = *(const v8h*)(p);          // K = kb .. kb+7
      u.h[1] = *(const v8h*)(p + 16);     // K = kb+16 .. kb+23
      bfr[j] = u.v;
    }

#pragma unroll
    for (int i = 0; i < 4; ++i)
#pragma unroll
      for (int j = 0; j < 4; ++j)
        acc[i][j] = __builtin_amdgcn_wmma_f32_16x16x32_f16(
            /*neg_a=*/false, afr[i], /*neg_b=*/false, bfr[j],
            /*c_mod=*/(short)0, acc[i][j], /*reuse_a=*/false, /*reuse_b=*/false);

    if (ks + 1 < KSTEPS) wait_async0();         // my LDS writes for `nxt` have landed
    __syncthreads();                            // everyone's have
  }

  // Epilogue. C/D layout: lanes 0-15 -> M = v, lanes 16-31 -> M = v+8; N = lane%16.
  const int mhi = (lane >> 4) * 8;
#pragma unroll
  for (int j = 0; j < 4; ++j) {
    const int col = nb0 + wn + j * 16 + lh;
    const float al = alpha[col];
#pragma unroll
    for (int i = 0; i < 4; ++i) {
      const int rbase = mb0 + wm + i * 16 + mhi;
#pragma unroll
      for (int v = 0; v < 8; ++v) {
        const int row = rbase + v;
        out[(size_t)row * DOUT + col] = acc[i][j][v] * al * rowBeta[row];
      }
    }
  }
}

// ---------------- Host launcher ----------------
extern "C" void kernel_launch(void* const* d_in, const int* in_sizes, int n_in,
                              void* d_out, int out_size, void* d_ws, size_t ws_size,
                              hipStream_t stream) {
  (void)in_sizes; (void)n_in; (void)out_size; (void)ws_size;
  const float* x     = (const float*)d_in[0];
  const float* gamma = (const float*)d_in[1];
  const float* bbeta = (const float*)d_in[2];
  const float* W     = (const float*)d_in[3];
  const float* alpha = (const float*)d_in[4];
  float* out = (float*)d_out;

  // Workspace layout (~97 MB total)
  float* psum     = (float*)d_ws;                 // 32*DIN
  float* psq      = psum + 32 * DIN;              // 32*DIN
  float* colScale = psq + 32 * DIN;               // DIN
  float* colBias  = colScale + DIN;               // DIN
  float* rowBeta  = colBias + DIN;                // NROWS
  h16*   hb       = (h16*)(rowBeta + NROWS);      // NROWS*DIN  (16B-aligned offset)
  h16*   Wh       = hb + (size_t)NROWS * DIN;     // DOUT*DIN

  k_colstat_partial<<<dim3(DIN / 256, NROWS / 256), 256, 0, stream>>>(x, psum, psq);
  k_colstat_final<<<DIN / 256, 256, 0, stream>>>(psum, psq, gamma, bbeta, colScale, colBias);
  k_binarize<<<NROWS, 256, 0, stream>>>(x, colScale, colBias, hb, rowBeta);
  k_w2h<<<(unsigned)(((size_t)DOUT * DIN) / 1024), 256, 0, stream>>>(W, Wh);
  k_bin_gemm<<<dim3(DOUT / TN, NROWS / TM), 256, 0, stream>>>(hb, Wh, alpha, rowBeta, out);
}
